// PAM_Module_87960930222505
// MI455X (gfx1250) — matrix operand order repaired and checked by reading, not compiled, not run on hardware
//
#include <hip/hip_runtime.h>

#define B_DIM 4
#define C_DIM 256
#define C8_DIM 32
#define N_DIM 4096

typedef __attribute__((ext_vector_type(16))) __bf16 v16bf;
typedef __attribute__((ext_vector_type(8)))  __bf16 v8bf;
typedef __attribute__((ext_vector_type(8)))  float  v8f;
typedef __attribute__((ext_vector_type(4)))  unsigned int v4u;
typedef __attribute__((ext_vector_type(8)))  int    v8i_t;
typedef __attribute__((ext_vector_type(4)))  int    v4i_t;

union BF16x16 {
  v16bf  v;
  v8bf   h[2];
  __bf16 e[16];
};

static __device__ __forceinline__ v8f wmma_bf16(const BF16x16& a, const BF16x16& b, v8f c) {
  // D = A(16x32 bf16) * B(32x16 bf16) + C(16x16 f32)
  return __builtin_amdgcn_wmma_f32_16x16x32_bf16(false, a.v, false, b.v, (short)0, c, false, false);
}

// Workgroup-relative LDS byte offset of a shared-memory pointer.
static __device__ __forceinline__ unsigned lds_offset(const void* p) {
  typedef __attribute__((address_space(3))) const char lds_char;
  return (unsigned)(unsigned long long)(lds_char*)p;
}

// ---------------------------------------------------------------------------
// Tensor Data Mover: DMA a 2-D bf16 tile (tile_w x tile_h elems, row stride
// row_stride_elems) from global memory into LDS at lds_byte_off.
// D# layout per CDNA5 ISA (groups 2/3 zero: <=2D tensor).
// This toolchain exposes the 6-arg builtin form:
//   (uint32x4 g0, int32x8 g1, int32x4, int32x4, int32x8, i32 cpol)
// ---------------------------------------------------------------------------
static __device__ __forceinline__ void tdm_load_2d_bf16(
    const __bf16* gsrc, unsigned lds_byte_off,
    unsigned tile_w, unsigned tile_h, unsigned row_stride_elems) {
  const unsigned long long ga = (unsigned long long)(uintptr_t)gsrc;
  v4u g0;
  g0[0] = 1u;                                                  // count=1, user desc
  g0[1] = lds_byte_off;                                        // lds_addr (bytes)
  g0[2] = (unsigned)ga;                                        // global_addr[31:0]
  g0[3] = (unsigned)((ga >> 32) & 0x01FFFFFFu) | (2u << 30);   // global_addr[56:32], type=2
  v8i_t g1;
  g1[0] = (int)(1u << 16);                                     // data_size=1 -> 2 bytes
  g1[1] = (int)((row_stride_elems & 0xFFFFu) << 16);           // tensor_dim0[15:0]
  g1[2] = (int)(((row_stride_elems >> 16) & 0xFFFFu) |         // tensor_dim0[31:16]
                ((tile_h & 0xFFFFu) << 16));                   // tensor_dim1[15:0]
  g1[3] = (int)(((tile_h >> 16) & 0xFFFFu) |                   // tensor_dim1[31:16]
                ((tile_w & 0xFFFFu) << 16));                   // tile_dim0
  g1[4] = (int)(tile_h & 0xFFFFu);                             // tile_dim1 (tile_dim2=0)
  g1[5] = (int)row_stride_elems;                               // tensor_dim0_stride[31:0]
  g1[6] = 0;                                                   // stride[47:32], dim1_stride
  g1[7] = 0;
  const v4i_t gz4 = {0, 0, 0, 0};
  const v8i_t gz8 = {0, 0, 0, 0, 0, 0, 0, 0};
  __builtin_amdgcn_tensor_load_to_lds(g0, g1, gz4, gz4, gz8, 0);
}

// ---------------------------------------------------------------------------
// Stage 0: fold depthwise conv into pointwise bias:
//   bias'[o] = pb[o] + sum_c pw[o,c]*db[c]
// ---------------------------------------------------------------------------
__global__ __launch_bounds__(320)
void s0_bias(const float* __restrict__ pwq, const float* __restrict__ pbq, const float* __restrict__ dbq,
             const float* __restrict__ pwk, const float* __restrict__ pbk, const float* __restrict__ dbk,
             const float* __restrict__ pwv, const float* __restrict__ pbv, const float* __restrict__ dbv,
             float* __restrict__ biasq, float* __restrict__ biask, float* __restrict__ biasv) {
  const int t = threadIdx.x;
  if (t < 32) {
    float s = pbq[t];
    for (int c = 0; c < C_DIM; ++c) s += pwq[t * C_DIM + c] * dbq[c];
    biasq[t] = s;
  } else if (t < 64) {
    const int o = t - 32;
    float s = pbk[o];
    for (int c = 0; c < C_DIM; ++c) s += pwk[o * C_DIM + c] * dbk[c];
    biask[o] = s;
  } else if (t < 320) {
    const int o = t - 64;
    float s = pbv[o];
    for (int c = 0; c < C_DIM; ++c) s += pwv[o * C_DIM + c] * dbv[c];
    biasv[o] = s;
  }
}

// ---------------------------------------------------------------------------
// Stage 1: q/k/v projections via WMMA.
//   q[b,o,n] = sum_c (pw_q[o,c]*dw_q[c]) * x[b,c,n] + bias'_q[o]   (o<32)
// Outputs:
//   qt,kt : bf16 [B][N][32]   (row n contiguous over c8 -> WMMA A/B frag loads)
//   vt    : bf16 [B][C][N]    (row c contiguous over j  -> PV B frag loads)
// One wave per (b, 16-wide n tile); x B-fragments reused across 20 o-tiles.
// ---------------------------------------------------------------------------
__global__ __launch_bounds__(256)
void s1_qkv(const float* __restrict__ x,
            const float* __restrict__ pwq, const float* __restrict__ dwq,
            const float* __restrict__ pwk, const float* __restrict__ dwk,
            const float* __restrict__ pwv, const float* __restrict__ dwv,
            const float* __restrict__ biasq, const float* __restrict__ biask,
            const float* __restrict__ biasv,
            __bf16* __restrict__ qt, __bf16* __restrict__ kt, __bf16* __restrict__ vt) {
  const int lane = threadIdx.x & 31;
  const int wave = threadIdx.x >> 5;
  const int wg   = blockIdx.x * 8 + wave;          // 0..1023
  const int b    = wg >> 8;
  const int n0   = (wg & 255) << 4;
  const int ln   = lane & 15;
  const int hi   = lane >> 4;
  const int n    = n0 + ln;

  // B fragments of x for all 8 K-chunks (K = 256 channels, 32 per chunk).
  BF16x16 xb[8];
#pragma unroll
  for (int kc = 0; kc < 8; ++kc) {
    const int cbase = kc * 32 + hi * 16;
    const float* xp = x + ((size_t)(b * C_DIM + cbase)) * N_DIM + n;
#pragma unroll
    for (int t = 0; t < 16; ++t) xb[kc].e[t] = (__bf16)xp[(size_t)t * N_DIM];
  }

  // A(16x32): lane -> row o; lanes 0-15 hold K {0..15}, 16-31 hold {16..31}.
  auto doTile = [&](const float* __restrict__ pw, const float* __restrict__ dw, int o0) -> v8f {
    v8f acc = {};
#pragma unroll
    for (int kc = 0; kc < 8; ++kc) {
      BF16x16 a;
      const int g0 = kc * 32 + hi * 16;
      const float* wp = pw + (size_t)(o0 + ln) * C_DIM + g0;
#pragma unroll
      for (int t = 0; t < 8; ++t) a.e[t] = (__bf16)(wp[t] * dw[g0 + t]);
#pragma unroll
      for (int t = 0; t < 8; ++t) a.e[t + 8] = (__bf16)(wp[8 + t] * dw[g0 + 8 + t]);
      acc = wmma_bf16(a, xb[kc], acc);
    }
    return acc;
  };

  // q and k: O = 32 -> 2 o-tiles each, stored transposed [B][N][32].
  {
    v8f q0 = doTile(pwq, dwq, 0);
    v8f q1 = doTile(pwq, dwq, 16);
    v8f k0 = doTile(pwk, dwk, 0);
    v8f k1 = doTile(pwk, dwk, 16);
    __bf16* qrow = qt + ((size_t)b * N_DIM + n) * C8_DIM;
    __bf16* krow = kt + ((size_t)b * N_DIM + n) * C8_DIM;
#pragma unroll
    for (int r = 0; r < 8; ++r) {
      const int o = r + 8 * hi;
      qrow[o]      = (__bf16)(q0[r] + biasq[o]);
      qrow[o + 16] = (__bf16)(q1[r] + biasq[o + 16]);
      krow[o]      = (__bf16)(k0[r] + biask[o]);
      krow[o + 16] = (__bf16)(k1[r] + biask[o + 16]);
    }
  }

  // v: O = 256 -> 16 o-tiles, stored [B][C][N].
  for (int ot = 0; ot < 16; ++ot) {
    v8f acc = doTile(pwv, dwv, ot * 16);
#pragma unroll
    for (int r = 0; r < 8; ++r) {
      const int o = ot * 16 + r + 8 * hi;
      vt[((size_t)(b * C_DIM + o)) * N_DIM + n] = (__bf16)(acc[r] + biasv[o]);
    }
  }
}

// ---------------------------------------------------------------------------
// Stage 2: fused flash attention + residual.
// Block = 4 waves sharing one batch; each wave owns a 16-row query tile.
// Per 32-key chunk: TDM double-buffers the K chunk (2KB) and V chunk (16KB)
// into LDS (DMA of chunk j+1 overlaps compute of chunk j), then
//   2 QK^T WMMAs -> online softmax (shfl reductions within 16-lane halves)
//   -> P transpose via per-wave LDS tile -> 16 PV WMMAs into O[16x256].
// ---------------------------------------------------------------------------
#define S2_WAVES 4
struct S2Shared {
  __bf16 vbuf[2][C_DIM * 32];      // 2 x 16KB : V chunk  [c][32 j]
  __bf16 kbuf[2][32 * 32];         // 2 x  2KB : K chunk  [j][32 c8]
  __bf16 plds[S2_WAVES][16 * 32];  // 4 x  1KB : P transpose tiles
};

__global__ __launch_bounds__(32 * S2_WAVES)
void s2_attn(const __bf16* __restrict__ qt, const __bf16* __restrict__ kt,
             const __bf16* __restrict__ vt, const float* __restrict__ xin,
             const float* __restrict__ gammap, float* __restrict__ outp) {
  __shared__ __align__(16) S2Shared smem;

  const int lane = threadIdx.x & 31;
  const int wave = threadIdx.x >> 5;
  const int wg   = blockIdx.x * S2_WAVES + wave;    // 0..1023
  const int b    = wg >> 8;                         // same for all waves in block
  const int i0   = (wg & 255) << 4;
  const int ln   = lane & 15;
  const int hi   = lane >> 4;

  // Q A-fragment: row i = i0+ln, K = c8 (32), groups per half-wave.
  BF16x16 qa;
  {
    const __bf16* qrow = qt + ((size_t)b * N_DIM + i0 + ln) * C8_DIM + hi * 16;
    qa.h[0] = *(const v8bf*)(qrow);
    qa.h[1] = *(const v8bf*)(qrow + 8);
  }

  v8f o_acc[16];
#pragma unroll
  for (int t = 0; t < 16; ++t) o_acc[t] = (v8f){};
  float m[8], l[8];
#pragma unroll
  for (int r = 0; r < 8; ++r) { m[r] = -3.0e38f; l[r] = 0.0f; }

  __bf16* pl = smem.plds[wave];

  // TDM issue for key/value chunk jc into LDS buffer `buf` (wave 0 only).
  auto issue_chunk = [&](int jc, int buf) {
    const int j0 = jc * 32;
    // K chunk: 32 rows of kt are fully contiguous (32*32 bf16) -> 1-D tile.
    tdm_load_2d_bf16(kt + ((size_t)b * N_DIM + j0) * C8_DIM,
                     lds_offset(&smem.kbuf[buf][0]),
                     /*tile_w=*/1024, /*tile_h=*/1, /*row_stride=*/1024);
    // V chunk: 256 rows (c) x 32 elems (j), row stride N.
    tdm_load_2d_bf16(vt + ((size_t)b * C_DIM) * N_DIM + j0,
                     lds_offset(&smem.vbuf[buf][0]),
                     /*tile_w=*/32, /*tile_h=*/C_DIM, /*row_stride=*/N_DIM);
  };

  const int NCH = N_DIM / 32;
  if (wave == 0) issue_chunk(0, 0);

  for (int jc = 0; jc < NCH; ++jc) {
    const int buf = jc & 1;
    if (wave == 0) __builtin_amdgcn_s_wait_tensorcnt(0);   // chunk jc landed
    __syncthreads();                                       // publish to all waves
    if (wave == 0 && jc + 1 < NCH) issue_chunk(jc + 1, buf ^ 1);  // overlap DMA

    // K B-fragments for the two 16-key subtiles, from LDS.
    BF16x16 kb0, kb1;
    const __bf16* kr0 = &smem.kbuf[buf][ln * 32 + hi * 16];
    kb0.h[0] = *(const v8bf*)(kr0);
    kb0.h[1] = *(const v8bf*)(kr0 + 8);
    const __bf16* kr1 = kr0 + 16 * 32;
    kb1.h[0] = *(const v8bf*)(kr1);
    kb1.h[1] = *(const v8bf*)(kr1 + 8);

    v8f s0 = {}, s1 = {};
    s0 = wmma_bf16(qa, kb0, s0);
    s1 = wmma_bf16(qa, kb1, s1);

    // Online softmax per row. C-layout: vgpr r holds row r+8*hi across a
    // 16-lane half, so xor-masks 1/2/4/8 reduce exactly one row.
    float alpha[8];
#pragma unroll
    for (int r = 0; r < 8; ++r) {
      float a0 = s0[r], a1 = s1[r];
      float mloc = fmaxf(a0, a1);
      mloc = fmaxf(mloc, __shfl_xor(mloc, 1, 32));
      mloc = fmaxf(mloc, __shfl_xor(mloc, 2, 32));
      mloc = fmaxf(mloc, __shfl_xor(mloc, 4, 32));
      mloc = fmaxf(mloc, __shfl_xor(mloc, 8, 32));
      const float mn = fmaxf(m[r], mloc);
      const float al = __expf(m[r] - mn);
      const float e0 = __expf(a0 - mn);
      const float e1 = __expf(a1 - mn);
      float rs = e0 + e1;
      rs += __shfl_xor(rs, 1, 32);
      rs += __shfl_xor(rs, 2, 32);
      rs += __shfl_xor(rs, 4, 32);
      rs += __shfl_xor(rs, 8, 32);
      l[r] = l[r] * al + rs;
      m[r] = mn;
      alpha[r] = al;
      // Stash P into LDS in row-major [i][j] for the A-fragment reload.
      const int mrow = r + 8 * hi;
      pl[mrow * 32 + ln]      = (__bf16)e0;
      pl[mrow * 32 + ln + 16] = (__bf16)e1;
    }

    // Rescale accumulators by alpha (same row<->vgpr mapping as S).
#pragma unroll
    for (int ct = 0; ct < 16; ++ct) {
#pragma unroll
      for (int r = 0; r < 8; ++r) o_acc[ct][r] *= alpha[r];
    }

    // DS ops from one wave complete in order; just stop the compiler moving
    // the transposed reload above the stores.
    __builtin_amdgcn_wave_barrier();
    asm volatile("" ::: "memory");

    BF16x16 pa;
    const __bf16* prow = pl + ln * 32 + hi * 16;
    pa.h[0] = *(const v8bf*)(prow);
    pa.h[1] = *(const v8bf*)(prow + 8);

    __builtin_amdgcn_wave_barrier();
    asm volatile("" ::: "memory");

    // O[16 x 256] += P(16x32) * V^T(32x16) per 16-channel tile, V from LDS.
#pragma unroll
    for (int ct = 0; ct < 16; ++ct) {
      BF16x16 vb;
      const __bf16* vrow = &smem.vbuf[buf][(ct * 16 + ln) * 32 + hi * 16];
      vb.h[0] = *(const v8bf*)(vrow);
      vb.h[1] = *(const v8bf*)(vrow + 8);
      o_acc[ct] = wmma_bf16(pa, vb, o_acc[ct]);
    }
  }

  // Epilogue: out = gamma * (O / l) + x.
  const float g = gammap[0];
#pragma unroll
  for (int ct = 0; ct < 16; ++ct) {
    const int c = ct * 16 + ln;
#pragma unroll
    for (int r = 0; r < 8; ++r) {
      const int i = i0 + r + 8 * hi;
      const size_t idx = ((size_t)(b * C_DIM + c)) * N_DIM + i;
      outp[idx] = g * (o_acc[ct][r] / l[r]) + xin[idx];
    }
  }
}

// ---------------------------------------------------------------------------
extern "C" void kernel_launch(void* const* d_in, const int* in_sizes, int n_in,
                              void* d_out, int out_size, void* d_ws, size_t ws_size,
                              hipStream_t stream) {
  (void)in_sizes; (void)n_in; (void)out_size; (void)ws_size;

  const float* x     = (const float*)d_in[0];
  const float* dw_q  = (const float*)d_in[1];
  const float* db_q  = (const float*)d_in[2];
  const float* pw_q  = (const float*)d_in[3];
  const float* pb_q  = (const float*)d_in[4];
  const float* dw_k  = (const float*)d_in[5];
  const float* db_k  = (const float*)d_in[6];
  const float* pw_k  = (const float*)d_in[7];
  const float* pb_k  = (const float*)d_in[8];
  const float* dw_v  = (const float*)d_in[9];
  const float* db_v  = (const float*)d_in[10];
  const float* pw_v  = (const float*)d_in[11];
  const float* pb_v  = (const float*)d_in[12];
  const float* gamma = (const float*)d_in[13];
  float* out = (float*)d_out;

  // Workspace layout (~10 MB):
  char* ws = (char*)d_ws;
  __bf16* qt = (__bf16*)(ws);                       // 1 MB : [B][N][32]
  __bf16* kt = (__bf16*)(ws + (1u << 20));          // 1 MB : [B][N][32]
  __bf16* vt = (__bf16*)(ws + (2u << 20));          // 8 MB : [B][C][N]
  float*  biasq = (float*)(ws + (10u << 20));       // 32 f
  float*  biask = biasq + 32;                       // 32 f
  float*  biasv = biasq + 64;                       // 256 f

  s0_bias<<<1, 320, 0, stream>>>(pw_q, pb_q, db_q, pw_k, pb_k, db_k,
                                 pw_v, pb_v, db_v, biasq, biask, biasv);

  // B*(N/16) = 1024 waves, 8 waves/block.
  s1_qkv<<<128, 256, 0, stream>>>(x, pw_q, dw_q, pw_k, dw_k, pw_v, dw_v,
                                  biasq, biask, biasv, qt, kt, vt);

  // B*(N/16) = 1024 waves, 4 waves/block, one batch per block.
  s2_attn<<<256, 32 * S2_WAVES, 0, stream>>>(qt, kt, vt, x, gamma, out);
}
